// MambaAttentionBlock_10857677324838
// MI455X (gfx1250) — matrix operand order repaired and checked by reading, not compiled, not run on hardware
//
#include <hip/hip_runtime.h>

// ---------------------------------------------------------------------------
// MambaAttentionBlock for MI455X (gfx1250, wave32, WMMA + TDM).
// GEMMs: v_wmma_f32_16x16x32_bf16, operands staged to LDS with the Tensor
// Data Mover (tensor_load_to_lds, TENSORcnt), double-buffered so DMA of
// slab i+1 overlaps WMMA on slab i. TDM OOB-zero handles the N=56 x-proj.
// Selective scan is chunk-parallelized (16 x 144) with analytic correction.
// ---------------------------------------------------------------------------

#define DIMC   384
#define NHH    8
#define NST    16
#define KCONV  4
#define DINc   768
#define DTRc   24
#define BB     8
#define LTOK   2304
#define MTOK   (BB * LTOK)     // 18432 token rows
#define NCHUNK 16
#define CLEN   (LTOK / NCHUNK) // 144

#define BLK_M  128
#define BLK_N  64
#define BLK_K  32
#define LDA_SH 40              // 32 bf16 row + 8 bf16 (16B) TDM pad -> 80B stride

typedef __attribute__((ext_vector_type(16))) __bf16 v16bf;
typedef __attribute__((ext_vector_type(8)))  __bf16 v8bf;
typedef __attribute__((ext_vector_type(8)))  float  v8f;
typedef __attribute__((ext_vector_type(4)))  unsigned int v4u;
typedef __attribute__((ext_vector_type(4)))  int    v4i;
typedef __attribute__((ext_vector_type(8)))  int    v8i;

// ------------------------------- helpers -----------------------------------

__device__ __forceinline__ unsigned short f2bf(float f) {
  unsigned u = __float_as_uint(f);
  unsigned r = u + 0x7FFFu + ((u >> 16) & 1u);   // round-to-nearest-even
  return (unsigned short)(r >> 16);
}

__device__ __forceinline__ float silu_f(float x) { return x / (1.f + __expf(-x)); }

__device__ __forceinline__ float softplus_f(float x) {
  return (x > 20.f) ? x : log1pf(__expf(x));
}

__device__ __forceinline__ float gelu_f(float x) {  // tanh approximation (jax default)
  float x3 = x * x * x;
  return 0.5f * x * (1.f + tanhf(0.7978845608028654f * (x + 0.044715f * x3)));
}

__device__ __forceinline__ float wave_sum(float v) {
#pragma unroll
  for (int o = 16; o > 0; o >>= 1) v += __shfl_down(v, o);
  return v;
}
__device__ __forceinline__ float wave_max(float v) {
#pragma unroll
  for (int o = 16; o > 0; o >>= 1) v = fmaxf(v, __shfl_down(v, o));
  return v;
}

// --------------------------- TDM 2-D tile load ------------------------------
// Loads a [tile_rows x 32] bf16 tile (row stride = stride_elems in memory)
// into LDS at lds_addr with a 16-byte pad after every 64-byte row
// (pad_interval=16 DWORDs, pad_amount=4 DWORDs -> LDS row stride 80B).
// Rows beyond rows_rem read as zero (hardware OOB fill).
__device__ __forceinline__ void tdm_load_tile(unsigned lds_addr,
                                              const unsigned short* gptr,
                                              int td0_elems, int rows_rem,
                                              int tile_rows, int stride_elems) {
  unsigned long long ga = (unsigned long long)(uintptr_t)gptr;
  v4u g0;
  g0[0] = 1u;                                   // count=1, user descriptor
  g0[1] = lds_addr;                             // LDS byte address
  g0[2] = (unsigned)(ga & 0xFFFFFFFFu);         // global_addr[31:0]
  g0[3] = (unsigned)((ga >> 32) & 0x1FFFFFFu) | (2u << 30);  // [56:32] | type=2
  v8i g1;
  // wgmask=0 | data_size=1 (2B) | pad_enable | pad_interval=3 (16 dw) | pad_amount=3 (4 dw)
  g1[0] = (int)((1u << 16) | (1u << 20) | (3u << 22) | (3u << 25));
  g1[1] = (int)(((unsigned)td0_elems & 0xFFFFu) << 16);          // tensor_dim0 lo
  g1[2] = (int)((((unsigned)td0_elems >> 16) & 0xFFFFu) |
                (((unsigned)rows_rem & 0xFFFFu) << 16));         // td0 hi | td1 lo
  g1[3] = (int)((((unsigned)rows_rem >> 16) & 0xFFFFu) |
                ((unsigned)BLK_K << 16));                        // td1 hi | tile_dim0
  g1[4] = (int)((unsigned)tile_rows & 0xFFFFu);                  // tile_dim1 (tile_dim2=0)
  g1[5] = (int)(unsigned)stride_elems;                           // tensor_dim0_stride lo
  g1[6] = 0;                                                     // stride0 hi | stride1 lo
  g1[7] = 0;
  v4i z4 = {0, 0, 0, 0};
  v8i z8 = {0, 0, 0, 0, 0, 0, 0, 0};
  __builtin_amdgcn_tensor_load_to_lds(g0, g1, z4, z4, z8, 0);
}

// ---------------------- WMMA fragment loads (from LDS) ----------------------
// A tile in LDS: [rows][LDA_SH] bf16.  lane m<16: K {0..7,16..23}; lane>=16: +8.
__device__ __forceinline__ v16bf lds_frag_a(const unsigned short* sh, int row0,
                                            int lane) {
  int row  = row0 + (lane & 15);
  int koff = (lane < 16) ? 0 : 8;
  const unsigned short* p = sh + row * LDA_SH + koff;
  union { v16bf v; v8bf h[2]; } u;
  u.h[0] = *reinterpret_cast<const v8bf*>(p);
  u.h[1] = *reinterpret_cast<const v8bf*>(p + 16);
  return u.v;
}
// W tile in LDS: [rows][LDA_SH] bf16. lane n<16: K {0..7,16..23}; lane>=16: +8.
__device__ __forceinline__ v16bf lds_frag_b(const unsigned short* sh, int n0,
                                            int lane) {
  int n    = n0 + (lane & 15);
  int koff = (lane < 16) ? 0 : 8;
  const unsigned short* p = sh + n * LDA_SH + koff;
  union { v16bf v; v8bf h[2]; } u;
  u.h[0] = *reinterpret_cast<const v8bf*>(p);
  u.h[1] = *reinterpret_cast<const v8bf*>(p + 16);
  return u.v;
}

__device__ __forceinline__ v8f wmma_bf16(v16bf a, v16bf b, v8f c) {
  return __builtin_amdgcn_wmma_f32_16x16x32_bf16(false, a, false, b, (short)0, c,
                                                 false, false);
}

// ------------------------------ GEMM ---------------------------------------
// C(M,N) = A(M,K) @ W(N,K)^T (+bias) with fused epilogue.
// EPI 0: f32 store (+bias)            EPI 1: softplus -> f32 (delta)
// EPI 2: xs += (acc+b)*(1+mod[b,n])   EPI 3: gelu -> bf16
// EPI 4: out[b,n,l] = xs + acc + b    (final transposed store)
template <int EPI>
__device__ __forceinline__ void epi_store(int m, int n, int N, float acc,
                                          const float* __restrict__ bias,
                                          float* __restrict__ Cf,
                                          unsigned short* __restrict__ Cbf,
                                          float* __restrict__ xs,
                                          const float* __restrict__ mod,
                                          float* __restrict__ outp, int ldc) {
  if (n >= N) return;
  float bv = bias ? bias[n] : 0.f;
  if (EPI == 0) {
    Cf[(size_t)m * ldc + n] = acc + bv;
  } else if (EPI == 1) {
    Cf[(size_t)m * ldc + n] = softplus_f(acc + bv);
  } else if (EPI == 2) {
    int b = m / LTOK;
    xs[(size_t)m * DIMC + n] += (acc + bv) * (1.f + mod[b * DIMC + n]);
  } else if (EPI == 3) {
    Cbf[(size_t)m * ldc + n] = f2bf(gelu_f(acc + bv));
  } else {
    int b = m / LTOK, l = m - b * LTOK;
    outp[((size_t)b * DIMC + n) * LTOK + l] = xs[(size_t)m * DIMC + n] + acc + bv;
  }
}

template <int EPI>
__global__ __launch_bounds__(256) void gemm_bf16(
    const unsigned short* __restrict__ A, const unsigned short* __restrict__ W,
    int M, int N, int K, const float* __restrict__ bias, float* __restrict__ Cf,
    unsigned short* __restrict__ Cbf, float* __restrict__ xs,
    const float* __restrict__ mod, float* __restrict__ outp, int ldc) {
  __shared__ unsigned short shA[2][BLK_M * LDA_SH];
  __shared__ unsigned short shW[2][BLK_N * LDA_SH];

  int lane = threadIdx.x & 31;
  int w    = threadIdx.x >> 5;          // 8 waves: 4 (M) x 2 (N)
  int wm   = w >> 1, wn = w & 1;
  int m0b  = blockIdx.x * BLK_M;
  int n0b  = blockIdx.y * BLK_N;
  bool issuer = (w == 0);

  int nslab = K / BLK_K;
  // prologue: DMA slab 0 into buffer 0 (wave 0 issues, EXEC-independent)
  if (issuer) {
    tdm_load_tile((unsigned)(uintptr_t)&shA[0][0], A + (size_t)m0b * K,
                  K, M - m0b, BLK_M, K);
    tdm_load_tile((unsigned)(uintptr_t)&shW[0][0], W + (size_t)n0b * K,
                  K, N - n0b, BLK_N, K);
  }

  v8f acc[2][2] = {{{}, {}}, {{}, {}}};
  for (int i = 0; i < nslab; ++i) {
    int buf = i & 1;
    __builtin_amdgcn_s_wait_tensorcnt(0);   // slab i resident (issuer's cnt)
    __syncthreads();                        // visible to all waves; prev reads done
    if (issuer && (i + 1) < nslab) {        // overlap DMA of slab i+1 with math
      int k0 = (i + 1) * BLK_K;
      tdm_load_tile((unsigned)(uintptr_t)&shA[buf ^ 1][0],
                    A + (size_t)m0b * K + k0, K, M - m0b, BLK_M, K);
      tdm_load_tile((unsigned)(uintptr_t)&shW[buf ^ 1][0],
                    W + (size_t)n0b * K + k0, K, N - n0b, BLK_N, K);
    }
    v16bf a0 = lds_frag_a(&shA[buf][0], wm * 32, lane);
    v16bf a1 = lds_frag_a(&shA[buf][0], wm * 32 + 16, lane);
    v16bf b0 = lds_frag_b(&shW[buf][0], wn * 32, lane);
    v16bf b1 = lds_frag_b(&shW[buf][0], wn * 32 + 16, lane);
    acc[0][0] = wmma_bf16(a0, b0, acc[0][0]);
    acc[0][1] = wmma_bf16(a0, b1, acc[0][1]);
    acc[1][0] = wmma_bf16(a1, b0, acc[1][0]);
    acc[1][1] = wmma_bf16(a1, b1, acc[1][1]);
  }

  int mseg = (lane >> 4) << 3;            // +0 (lanes 0-15) / +8 (lanes 16-31)
  int ncl  = lane & 15;
#pragma unroll
  for (int mi = 0; mi < 2; ++mi) {
#pragma unroll
    for (int nj = 0; nj < 2; ++nj) {
#pragma unroll
      for (int r = 0; r < 8; ++r) {
        int m = m0b + wm * 32 + mi * 16 + mseg + r;
        int n = n0b + wn * 32 + nj * 16 + ncl;
        epi_store<EPI>(m, n, N, acc[mi][nj][r], bias, Cf, Cbf, xs, mod, outp, ldc);
      }
    }
  }
}

// ------------------- stage 0: transpose + LN1 + pool logits ----------------
__global__ __launch_bounds__(128) void k_trans_ln1_logits(
    const float* __restrict__ x, const float* __restrict__ g,
    const float* __restrict__ bta, const float* __restrict__ ap_w,
    const float* __restrict__ ap_b, float* __restrict__ xs,
    float* __restrict__ xn, float* __restrict__ logits) {
  int m = blockIdx.x;            // token row = b*LTOK + l
  int b = m / LTOK, l = m - b * LTOK;
  int tid = threadIdx.x, lane = tid & 31, wid = tid >> 5;
  __shared__ float sm[8];
  __shared__ float xnsh[DIMC];

  float v[3];
#pragma unroll
  for (int j = 0; j < 3; ++j) {
    int c = tid + j * 128;
    v[j]  = x[((size_t)b * DIMC + c) * LTOK + l];
  }
  float s = wave_sum(v[0] + v[1] + v[2]);
  if (lane == 0) sm[wid] = s;
  __syncthreads();
  float mean = (sm[0] + sm[1] + sm[2] + sm[3]) * (1.f / DIMC);
  __syncthreads();
  float q = 0.f;
#pragma unroll
  for (int j = 0; j < 3; ++j) { float d = v[j] - mean; q += d * d; }
  q = wave_sum(q);
  if (lane == 0) sm[wid] = q;
  __syncthreads();
  float rstd = rsqrtf((sm[0] + sm[1] + sm[2] + sm[3]) * (1.f / DIMC) + 1e-5f);
#pragma unroll
  for (int j = 0; j < 3; ++j) {
    int c = tid + j * 128;
    float nv = (v[j] - mean) * rstd * g[c] + bta[c];
    xs[(size_t)m * DIMC + c] = v[j];
    xn[(size_t)m * DIMC + c] = nv;
    xnsh[c] = nv;
  }
  __syncthreads();
  // 8 heads x 16 lanes each: logits[b, head, l]
  int head = tid >> 4, sub = tid & 15;
  float p = 0.f;
#pragma unroll
  for (int j = 0; j < DIMC / 16; ++j) {
    int c = sub + j * 16;
    p += ap_w[head * DIMC + c] * xnsh[c];
  }
  p += __shfl_xor(p, 1); p += __shfl_xor(p, 2);
  p += __shfl_xor(p, 4); p += __shfl_xor(p, 8);
  if (sub == 0) logits[((size_t)b * NHH + head) * LTOK + l] = p + ap_b[head];
}

// softmax over L per (b, head), in place
__global__ __launch_bounds__(256) void k_softmax_L(float* __restrict__ logits) {
  float* row = logits + (size_t)blockIdx.x * LTOK;
  int tid = threadIdx.x, lane = tid & 31, wid = tid >> 5;
  __shared__ float sm[8];
  float mx = -3.4e38f;
  for (int i = tid; i < LTOK; i += 256) mx = fmaxf(mx, row[i]);
  mx = wave_max(mx);
  if (lane == 0) sm[wid] = mx;
  __syncthreads();
  float gmx = sm[0];
#pragma unroll
  for (int i = 1; i < 8; ++i) gmx = fmaxf(gmx, sm[i]);
  __syncthreads();
  float s = 0.f;
  for (int i = tid; i < LTOK; i += 256) s += __expf(row[i] - gmx);
  s = wave_sum(s);
  if (lane == 0) sm[wid] = s;
  __syncthreads();
  float tot = sm[0] + sm[1] + sm[2] + sm[3] + sm[4] + sm[5] + sm[6] + sm[7];
  float inv = 1.f / tot;
  for (int i = tid; i < LTOK; i += 256) row[i] = __expf(row[i] - gmx) * inv;
}

__global__ void k_wmean(const float* __restrict__ attn, float* __restrict__ wv) {
  int i = blockIdx.x * 256 + threadIdx.x;  // i = b*LTOK + l
  if (i >= BB * LTOK) return;
  int b = i / LTOK, l = i - b * LTOK;
  float s = 0.f;
#pragma unroll
  for (int n = 0; n < NHH; ++n) s += attn[((size_t)b * NHH + n) * LTOK + l];
  wv[i] = s * (1.f / NHH);
}

__global__ __launch_bounds__(128) void k_gc(const float* __restrict__ wv,
                                            const float* __restrict__ xn,
                                            float* __restrict__ gc) {
  int c = blockIdx.x * 128 + threadIdx.x;
  int b = blockIdx.y;
  float acc = 0.f;
  for (int l = 0; l < LTOK; ++l)
    acc += wv[b * LTOK + l] * xn[((size_t)b * LTOK + l) * DIMC + c];
  gc[b * DIMC + c] = acc;
}

__global__ __launch_bounds__(768) void k_mod(const float* __restrict__ gc,
                                             const float* __restrict__ w1,
                                             const float* __restrict__ b1,
                                             const float* __restrict__ w2,
                                             const float* __restrict__ b2,
                                             float* __restrict__ modv) {
  int b = blockIdx.x, tid = threadIdx.x;
  __shared__ float gcs[DIMC];
  __shared__ float h1[2 * DIMC];
  if (tid < DIMC) gcs[tid] = gc[b * DIMC + tid];
  __syncthreads();
  float s = b1[tid];
  for (int c = 0; c < DIMC; ++c) s += w1[tid * DIMC + c] * gcs[c];
  h1[tid] = silu_f(s);
  __syncthreads();
  if (tid < DIMC) {
    float o = b2[tid];
    for (int j = 0; j < 2 * DIMC; ++j) o += w2[tid * 2 * DIMC + j] * h1[j];
    modv[b * DIMC + tid] = o;
  }
}

// generic LayerNorm over 384 -> bf16 output
__global__ __launch_bounds__(128) void k_ln_bf16(const float* __restrict__ X,
                                                 const float* __restrict__ g,
                                                 const float* __restrict__ bta,
                                                 unsigned short* __restrict__ out) {
  int m = blockIdx.x, tid = threadIdx.x, lane = tid & 31, wid = tid >> 5;
  __shared__ float sm[8];
  const float* xr = X + (size_t)m * DIMC;
  float v[3];
#pragma unroll
  for (int j = 0; j < 3; ++j) v[j] = xr[tid + j * 128];
  float s = wave_sum(v[0] + v[1] + v[2]);
  if (lane == 0) sm[wid] = s;
  __syncthreads();
  float mean = (sm[0] + sm[1] + sm[2] + sm[3]) * (1.f / DIMC);
  __syncthreads();
  float q = 0.f;
#pragma unroll
  for (int j = 0; j < 3; ++j) { float d = v[j] - mean; q += d * d; }
  q = wave_sum(q);
  if (lane == 0) sm[wid] = q;
  __syncthreads();
  float rstd = rsqrtf((sm[0] + sm[1] + sm[2] + sm[3]) * (1.f / DIMC) + 1e-5f);
#pragma unroll
  for (int j = 0; j < 3; ++j) {
    int c = tid + j * 128;
    out[(size_t)m * DIMC + c] = f2bf((v[j] - mean) * rstd * g[c] + bta[c]);
  }
}

// -------------------------- small conversions ------------------------------
__global__ void k_cvt_bf16(const float* __restrict__ s, unsigned short* __restrict__ d,
                           int n) {
  int i = blockIdx.x * 256 + threadIdx.x;
  if (i < n) d[i] = f2bf(s[i]);
}
__global__ void k_pack_dtw(const float* __restrict__ w, unsigned short* __restrict__ d) {
  int i = blockIdx.x * 256 + threadIdx.x;            // over 768*32
  if (i >= DINc * 32) return;
  int r = i >> 5, c = i & 31;
  d[i] = (c < DTRc) ? f2bf(w[r * DTRc + c]) : (unsigned short)0;
}
__global__ void k_pack_dbc(const float* __restrict__ dbc, unsigned short* __restrict__ d) {
  int i = blockIdx.x * 256 + threadIdx.x;            // over MTOK*32
  if (i >= MTOK * 32) return;
  int m = i >> 5, c = i & 31;
  d[i] = (c < DTRc) ? f2bf(dbc[(size_t)m * 56 + c]) : (unsigned short)0;
}
__global__ void k_Aexp(const float* __restrict__ A_log, float* __restrict__ Aexp) {
  int i = blockIdx.x * 256 + threadIdx.x;
  if (i < DINc * NST) Aexp[i] = -__expf(A_log[i]);
}

// depthwise causal conv (K=4) + SiLU; writes f32 and bf16 copies
__global__ void k_conv_silu(const float* __restrict__ xz, const float* __restrict__ cw,
                            const float* __restrict__ cb, float* __restrict__ u,
                            unsigned short* __restrict__ ubf) {
  int i = blockIdx.x * 256 + threadIdx.x;            // i = m*DINc + d
  if (i >= MTOK * DINc) return;
  int d = i % DINc, m = i / DINc;
  int b = m / LTOK, l = m - b * LTOK;
  float acc = cb[d];
#pragma unroll
  for (int k = 0; k < KCONV; ++k) {
    int lp = l - (KCONV - 1) + k;
    if (lp >= 0)
      acc += cw[d * KCONV + k] * xz[((size_t)b * LTOK + lp) * (2 * DINc) + d];
  }
  float uv = silu_f(acc);
  u[i]   = uv;
  ubf[i] = f2bf(uv);
}

// ----------------------- chunked selective scan ----------------------------
// pass 1: local scan within each chunk; lanes = 16 states x 16 channels
__global__ __launch_bounds__(256) void k_scan1(
    const float* __restrict__ delta, const float* __restrict__ u,
    const float* __restrict__ dbc, const float* __restrict__ Aexp,
    float* __restrict__ ybuf, float* __restrict__ Sbuf, float* __restrict__ hbuf) {
  int n = threadIdx.x & 15, dl = threadIdx.x >> 4;
  int d = blockIdx.x * 16 + dl;
  int ch = blockIdx.y, b = blockIdx.z;
  float Adn = Aexp[d * NST + n];
  float h = 0.f, S = 0.f;
  int lbase = ch * CLEN;
  for (int t = 0; t < CLEN; ++t) {
    int    m  = b * LTOK + lbase + t;
    float  dt = delta[(size_t)m * DINc + d];
    float  ut = u[(size_t)m * DINc + d];
    float  Bn = dbc[(size_t)m * 56 + DTRc + n];
    float  Cn = dbc[(size_t)m * 56 + DTRc + NST + n];
    S += dt;
    h = h * __expf(dt * Adn) + dt * ut * Bn;
    float yp = h * Cn;
    yp += __shfl_xor(yp, 1); yp += __shfl_xor(yp, 2);
    yp += __shfl_xor(yp, 4); yp += __shfl_xor(yp, 8);
    if (n == 0) {
      ybuf[(size_t)m * DINc + d] = yp;
      Sbuf[(size_t)m * DINc + d] = S;
    }
  }
  hbuf[(((size_t)b * NCHUNK + ch) * DINc + d) * NST + n] = h;
}

// pass 2: propagate chunk-entry states H0 (16 serial chunk steps)
__global__ void k_scan2(const float* __restrict__ hbuf, const float* __restrict__ Sbuf,
                        const float* __restrict__ Aexp, float* __restrict__ H0buf) {
  int i = blockIdx.x * 256 + threadIdx.x;            // over B*DIN*NST
  if (i >= BB * DINc * NST) return;
  int n = i & 15, d = (i >> 4) % DINc, b = i / (DINc * NST);
  float A = Aexp[d * NST + n];
  float H = 0.f;
  for (int ch = 0; ch < NCHUNK; ++ch) {
    size_t idx = (((size_t)b * NCHUNK + ch) * DINc + d) * NST + n;
    H0buf[idx] = H;
    float Send = Sbuf[((size_t)b * LTOK + ch * CLEN + CLEN - 1) * DINc + d];
    H = hbuf[idx] + __expf(A * Send) * H;
  }
}

// pass 3: y += C . (exp(A*S_t) * H0), + u*Dp, * silu(z)  -> bf16 for out GEMM
__global__ void k_scan3_gate(const float* __restrict__ ybuf,
                             const float* __restrict__ Sbuf,
                             const float* __restrict__ dbc,
                             const float* __restrict__ H0buf,
                             const float* __restrict__ Aexp,
                             const float* __restrict__ u,
                             const float* __restrict__ Dp,
                             const float* __restrict__ xz,
                             unsigned short* __restrict__ ybf) {
  int i = blockIdx.x * 256 + threadIdx.x;            // i = m*DINc + d
  if (i >= MTOK * DINc) return;
  int d = i % DINc, m = i / DINc;
  int b = m / LTOK, l = m - b * LTOK, ch = l / CLEN;
  float S = Sbuf[i];
  float y = ybuf[i];
  float corr = 0.f;
#pragma unroll
  for (int n = 0; n < NST; ++n) {
    float Cn = dbc[(size_t)m * 56 + DTRc + NST + n];
    float H0 = H0buf[(((size_t)b * NCHUNK + ch) * DINc + d) * NST + n];
    corr += Cn * __expf(Aexp[d * NST + n] * S) * H0;
  }
  y += corr;
  y += u[i] * Dp[d];
  float z = xz[(size_t)m * (2 * DINc) + DINc + d];
  y *= silu_f(z);
  ybf[i] = f2bf(y);
}

// ------------------------------ launch -------------------------------------
extern "C" void kernel_launch(void* const* d_in, const int* in_sizes, int n_in,
                              void* d_out, int out_size, void* d_ws, size_t ws_size,
                              hipStream_t stream) {
  const float* x      = (const float*)d_in[0];
  const float* n1_g   = (const float*)d_in[1];
  const float* n1_b   = (const float*)d_in[2];
  const float* ap_w   = (const float*)d_in[3];
  const float* ap_b   = (const float*)d_in[4];
  const float* n2_g   = (const float*)d_in[5];
  const float* n2_b   = (const float*)d_in[6];
  const float* in_w   = (const float*)d_in[7];
  const float* in_b   = (const float*)d_in[8];
  const float* conv_w = (const float*)d_in[9];
  const float* conv_b = (const float*)d_in[10];
  const float* xp_w   = (const float*)d_in[11];
  const float* dt_w   = (const float*)d_in[12];
  const float* dt_b   = (const float*)d_in[13];
  const float* A_log  = (const float*)d_in[14];
  const float* Dp     = (const float*)d_in[15];
  const float* out_w  = (const float*)d_in[16];
  const float* out_b  = (const float*)d_in[17];
  const float* mod_w1 = (const float*)d_in[18];
  const float* mod_b1 = (const float*)d_in[19];
  const float* mod_w2 = (const float*)d_in[20];
  const float* mod_b2 = (const float*)d_in[21];
  const float* n3_g   = (const float*)d_in[22];
  const float* n3_b   = (const float*)d_in[23];
  const float* fc1_w  = (const float*)d_in[24];
  const float* fc1_b  = (const float*)d_in[25];
  const float* fc2_w  = (const float*)d_in[26];
  const float* fc2_b  = (const float*)d_in[27];
  float* outp = (float*)d_out;

  // workspace carve (~600 MB)
  char* base = (char*)d_ws;
  size_t off = 0;
  auto alloc = [&](size_t bytes) -> void* {
    void* p = base + off;
    off = (off + bytes + 255) & ~(size_t)255;
    return p;
  };
  float* xs      = (float*)alloc((size_t)MTOK * DIMC * 4);
  float* xn      = (float*)alloc((size_t)MTOK * DIMC * 4);
  float* logits  = (float*)alloc((size_t)BB * NHH * LTOK * 4);
  float* wv      = (float*)alloc((size_t)BB * LTOK * 4);
  float* gc      = (float*)alloc((size_t)BB * DIMC * 4);
  float* modv    = (float*)alloc((size_t)BB * DIMC * 4);
  float* Aexp    = (float*)alloc((size_t)DINc * NST * 4);
  float* xz      = (float*)alloc((size_t)MTOK * 2 * DINc * 4);
  float* u       = (float*)alloc((size_t)MTOK * DINc * 4);
  float* dbc     = (float*)alloc((size_t)MTOK * 56 * 4);
  float* delta   = (float*)alloc((size_t)MTOK * DINc * 4);
  float* Sbuf    = (float*)alloc((size_t)MTOK * DINc * 4);
  float* ybuf    = (float*)alloc((size_t)MTOK * DINc * 4);
  float* hbuf    = (float*)alloc((size_t)BB * NCHUNK * DINc * NST * 4);
  float* H0buf   = (float*)alloc((size_t)BB * NCHUNK * DINc * NST * 4);
  unsigned short* xn2_bf = (unsigned short*)alloc((size_t)MTOK * DIMC * 2);
  unsigned short* u_bf   = (unsigned short*)alloc((size_t)MTOK * DINc * 2);
  unsigned short* dbc_bf = (unsigned short*)alloc((size_t)MTOK * 32 * 2);
  unsigned short* y_bf   = (unsigned short*)alloc((size_t)MTOK * DINc * 2);
  unsigned short* h3_bf  = (unsigned short*)alloc((size_t)MTOK * DIMC * 2);
  unsigned short* h4_bf  = (unsigned short*)alloc((size_t)MTOK * 4 * DIMC * 2);
  unsigned short* inw_bf = (unsigned short*)alloc((size_t)2 * DINc * DIMC * 2);
  unsigned short* xpw_bf = (unsigned short*)alloc((size_t)56 * DINc * 2);
  unsigned short* dtw_bf = (unsigned short*)alloc((size_t)DINc * 32 * 2);
  unsigned short* outw_bf = (unsigned short*)alloc((size_t)DIMC * DINc * 2);
  unsigned short* fc1_bf  = (unsigned short*)alloc((size_t)4 * DIMC * DIMC * 2);
  unsigned short* fc2_bf  = (unsigned short*)alloc((size_t)DIMC * 4 * DIMC * 2);

  auto cdiv = [](int a, int b) { return (a + b - 1) / b; };

  // weight conversions
  k_Aexp<<<cdiv(DINc * NST, 256), 256, 0, stream>>>(A_log, Aexp);
  k_cvt_bf16<<<cdiv(2 * DINc * DIMC, 256), 256, 0, stream>>>(in_w, inw_bf, 2 * DINc * DIMC);
  k_cvt_bf16<<<cdiv(56 * DINc, 256), 256, 0, stream>>>(xp_w, xpw_bf, 56 * DINc);
  k_cvt_bf16<<<cdiv(DIMC * DINc, 256), 256, 0, stream>>>(out_w, outw_bf, DIMC * DINc);
  k_cvt_bf16<<<cdiv(4 * DIMC * DIMC, 256), 256, 0, stream>>>(fc1_w, fc1_bf, 4 * DIMC * DIMC);
  k_cvt_bf16<<<cdiv(DIMC * 4 * DIMC, 256), 256, 0, stream>>>(fc2_w, fc2_bf, DIMC * 4 * DIMC);
  k_pack_dtw<<<cdiv(DINc * 32, 256), 256, 0, stream>>>(dt_w, dtw_bf);

  // stage 1: transpose + LN1 + pooling logits, softmax, gc, mod
  k_trans_ln1_logits<<<MTOK, 128, 0, stream>>>(x, n1_g, n1_b, ap_w, ap_b, xs, xn, logits);
  k_softmax_L<<<BB * NHH, 256, 0, stream>>>(logits);
  k_wmean<<<cdiv(BB * LTOK, 256), 256, 0, stream>>>(logits, wv);
  k_gc<<<dim3(DIMC / 128, BB), 128, 0, stream>>>(wv, xn, gc);
  k_mod<<<BB, 2 * DIMC, 0, stream>>>(gc, mod_w1, mod_b1, mod_w2, mod_b2, modv);

  // stage 2: Mamba
  k_ln_bf16<<<MTOK, 128, 0, stream>>>(xs, n2_g, n2_b, xn2_bf);
  gemm_bf16<0><<<dim3(MTOK / BLK_M, cdiv(2 * DINc, BLK_N)), 256, 0, stream>>>(
      xn2_bf, inw_bf, MTOK, 2 * DINc, DIMC, in_b, xz, nullptr, nullptr, nullptr,
      nullptr, 2 * DINc);
  k_conv_silu<<<cdiv(MTOK * DINc, 256), 256, 0, stream>>>(xz, conv_w, conv_b, u, u_bf);
  gemm_bf16<0><<<dim3(MTOK / BLK_M, 1), 256, 0, stream>>>(
      u_bf, xpw_bf, MTOK, 56, DINc, nullptr, dbc, nullptr, nullptr, nullptr,
      nullptr, 56);
  k_pack_dbc<<<cdiv(MTOK * 32, 256), 256, 0, stream>>>(dbc, dbc_bf);
  gemm_bf16<1><<<dim3(MTOK / BLK_M, cdiv(DINc, BLK_N)), 256, 0, stream>>>(
      dbc_bf, dtw_bf, MTOK, DINc, 32, dt_b, delta, nullptr, nullptr, nullptr,
      nullptr, DINc);
  k_scan1<<<dim3(DINc / 16, NCHUNK, BB), 256, 0, stream>>>(delta, u, dbc, Aexp,
                                                           ybuf, Sbuf, hbuf);
  k_scan2<<<cdiv(BB * DINc * NST, 256), 256, 0, stream>>>(hbuf, Sbuf, Aexp, H0buf);
  k_scan3_gate<<<cdiv(MTOK * DINc, 256), 256, 0, stream>>>(ybuf, Sbuf, dbc, H0buf,
                                                           Aexp, u, Dp, xz, y_bf);
  gemm_bf16<2><<<dim3(MTOK / BLK_M, cdiv(DIMC, BLK_N)), 256, 0, stream>>>(
      y_bf, outw_bf, MTOK, DIMC, DINc, out_b, nullptr, nullptr, xs, modv,
      nullptr, DIMC);

  // stage 3: MLP + final transposed store
  k_ln_bf16<<<MTOK, 128, 0, stream>>>(xs, n3_g, n3_b, h3_bf);
  gemm_bf16<3><<<dim3(MTOK / BLK_M, cdiv(4 * DIMC, BLK_N)), 256, 0, stream>>>(
      h3_bf, fc1_bf, MTOK, 4 * DIMC, DIMC, fc1_b, nullptr, h4_bf, nullptr,
      nullptr, nullptr, 4 * DIMC);
  gemm_bf16<4><<<dim3(MTOK / BLK_M, cdiv(DIMC, BLK_N)), 256, 0, stream>>>(
      h4_bf, fc2_bf, MTOK, DIMC, 4 * DIMC, fc2_b, nullptr, nullptr, xs, nullptr,
      outp, DIMC);
}
